// Attention_80333068304619
// MI455X (gfx1250) — matrix operand order repaired and checked by reading, not compiled, not run on hardware
//
#include <hip/hip_runtime.h>

// ---------------------------------------------------------------------------
// MHA block for MI455X (gfx1250): bf16 WMMA everywhere, flash-style attention,
// double-buffered LDS staging via CDNA5 async global->LDS copies (ASYNCcnt).
//   x[4,2048,1024] @ w_qkv[1024,3072] + b  ->  Q,K,V  (head-major, V transposed)
//   flash attention per (b,h) with online softmax
//   attn_out[8192,1024] @ w_proj + b_proj -> d_out (f32)
// ---------------------------------------------------------------------------

typedef __attribute__((ext_vector_type(16))) __bf16 v16bf;
typedef __attribute__((ext_vector_type(8)))  float  v8f;

#define BATCH 4
#define SEQ   2048
#define DIM_  1024
#define NH    16
#define HD    64
#define MTOT  (BATCH * SEQ)          // 8192 rows

// ---- CDNA5 async global->LDS copy (tracked by ASYNCcnt), with sync fallback
#if defined(__has_builtin)
#if __has_builtin(__builtin_amdgcn_global_load_async_to_lds_b128) && \
    __has_builtin(__builtin_amdgcn_s_wait_asynccnt)
#define USE_ASYNC_LDS 1
#endif
#endif

#define AS1 __attribute__((address_space(1)))
#define AS3 __attribute__((address_space(3)))
typedef int async_v4i __attribute__((vector_size(16)));   // matches builtin's pointee type

__device__ __forceinline__ void copy16_to_lds(__bf16* lds_dst, const __bf16* gsrc) {
#ifdef USE_ASYNC_LDS
  __builtin_amdgcn_global_load_async_to_lds_b128(
      (AS1 async_v4i*)gsrc, (AS3 async_v4i*)lds_dst, 0, 0);
#else
  *reinterpret_cast<uint4*>(lds_dst) = *reinterpret_cast<const uint4*>(gsrc);
#endif
}
__device__ __forceinline__ void copy_fence() {
#ifdef USE_ASYNC_LDS
  __builtin_amdgcn_s_wait_asynccnt(0);
#endif
}

__device__ __forceinline__ __bf16 to_bf16(float f) {
  unsigned u = __builtin_bit_cast(unsigned, f);
  unsigned r = u + 0x7FFFu + ((u >> 16) & 1u);   // round-to-nearest-even
  unsigned short h = (unsigned short)(r >> 16);
  return __builtin_bit_cast(__bf16, h);
}

// A-fragment (16x32 bf16, M x K) from row-major LDS [rows][ld].
// contiguous k order used here (B uses the same): lanes 0-15 -> M=lane, K=0..15; lanes 16-31 -> K=16..31.
__device__ __forceinline__ v16bf load_A16x32(const __bf16* s, int ld, int m0, int k0, int lane) {
  const int m  = m0 + (lane & 15);
  const int hf = (lane >> 4) & 1;
  const __bf16* p = s + m * ld + k0 + hf * 16;
  v16bf a;
#pragma unroll
  for (int i = 0; i < 16; i++) a[i] = p[i];
  return a;
}

// B-fragment (32x16 bf16, K x N) from B^T stored row-major [N][ld] in LDS.
// contiguous k order used here (same as A): lanes 0-15 -> N=lane, K=0..15; lanes 16-31 -> K=16..31.
__device__ __forceinline__ v16bf load_B32x16(const __bf16* s, int ld, int n0, int k0, int lane) {
  const int n  = n0 + (lane & 15);
  const int hf = (lane >> 4) & 1;
  const __bf16* p = s + n * ld + k0 + hf * 16;
  v16bf b;
#pragma unroll
  for (int i = 0; i < 16; i++) b[i] = p[i];
  return b;
}

// ---------------------------------------------------------------------------
__global__ void cvt_bf16_kernel(const float* __restrict__ in, __bf16* __restrict__ out, int n) {
  int i = blockIdx.x * blockDim.x + threadIdx.x;
  if (i < n) out[i] = to_bf16(in[i]);
}

// out[n*K + k] = in[k*N + n]  (transpose + convert, weights -> B^T bf16)
__global__ void cvt_T_bf16_kernel(const float* __restrict__ in, __bf16* __restrict__ out,
                                  int K, int N) {
  int i = blockIdx.x * blockDim.x + threadIdx.x;
  if (i < K * N) {
    int n = i / K, k = i - n * K;
    out[i] = to_bf16(in[(size_t)k * N + n]);
  }
}

// ---------------------------------------------------------------------------
// Shared GEMM core: 128x128 block tile, K-step 32, double-buffered async LDS.
// 8 waves; wave w owns M-stripe w*16 and all 8 N-subtiles (64 acc VGPRs).
template <typename EPILOGUE>
__device__ __forceinline__ void gemm_core(const __bf16* __restrict__ A,
                                          const __bf16* __restrict__ Bt,
                                          __bf16* sA, __bf16* sB,   // [2][128*32] each
                                          int m_base, int n_base, EPILOGUE epi) {
  const int tid  = threadIdx.x;
  const int lane = tid & 31;
  const int wave = tid >> 5;

  v8f zero = {0.f, 0.f, 0.f, 0.f, 0.f, 0.f, 0.f, 0.f};
  v8f acc[8];
#pragma unroll
  for (int i = 0; i < 8; i++) acc[i] = zero;

  // stage one 128x32 A tile and 128x32 B^T tile into buffer `buf`
  auto stage = [&](int k0, int buf) {
#pragma unroll
    for (int i = 0; i < 2; i++) {               // 512 16B chunks, 2 per thread
      int c = tid + i * 256;
      int r = c >> 2, cc = c & 3;
      copy16_to_lds(sA + buf * (128 * 32) + r * 32 + cc * 8,
                    A + (size_t)(m_base + r) * DIM_ + k0 + cc * 8);
    }
#pragma unroll
    for (int i = 0; i < 2; i++) {
      int c = tid + i * 256;
      int r = c >> 2, cc = c & 3;
      copy16_to_lds(sB + buf * (128 * 32) + r * 32 + cc * 8,
                    Bt + (size_t)(n_base + r) * DIM_ + k0 + cc * 8);
    }
  };

  stage(0, 0);
  copy_fence();
  __syncthreads();

  const int NK = DIM_ / 32;                     // 32 k-steps
  for (int t = 0; t < NK; t++) {
    const int buf = t & 1;
    if (t + 1 < NK) stage((t + 1) * 32, buf ^ 1);   // prefetch next tile (async)

    v16bf af = load_A16x32(sA + buf * (128 * 32), 32, wave * 16, 0, lane);
#pragma unroll
    for (int ns = 0; ns < 8; ns++) {
      v16bf bf_ = load_B32x16(sB + buf * (128 * 32), 32, ns * 16, 0, lane);
      acc[ns] = __builtin_amdgcn_wmma_f32_16x16x32_bf16(
          false, af, false, bf_, (short)0, acc[ns], false, false);
    }
    copy_fence();            // own async writes to buf^1 complete
    __syncthreads();         // all waves done reading buf / writing buf^1
  }
  epi(acc);
}

// QKV GEMM: C[8192,3072] = A @ B; epilogue scatters to Q (scaled), K, V^T.
__global__ __launch_bounds__(256) void qkv_gemm_kernel(
    const __bf16* __restrict__ A, const __bf16* __restrict__ Bt,
    const float* __restrict__ bias,
    __bf16* __restrict__ Qb, __bf16* __restrict__ Kb, __bf16* __restrict__ Vtb) {
  __shared__ alignas(16) __bf16 sA[2][128 * 32];
  __shared__ alignas(16) __bf16 sB[2][128 * 32];
  const int lane = threadIdx.x & 31;
  const int wave = threadIdx.x >> 5;
  const int hf   = (lane >> 4) & 1;
  const int l16  = lane & 15;
  const int m_base = blockIdx.x * 128;
  const int n_base = blockIdx.y * 128;

  gemm_core(A, Bt, &sA[0][0], &sB[0][0], m_base, n_base, [&](v8f* acc) {
#pragma unroll
    for (int ns = 0; ns < 8; ns++) {
      int ncol  = n_base + ns * 16 + l16;
      float bv  = bias[ncol];
      int which = ncol >> 10;        // 0=Q 1=K 2=V (tile never crosses a 1024 boundary)
      int d = ncol & 1023;
      int h = d >> 6;
      int e = d & 63;
#pragma unroll
      for (int v = 0; v < 8; v++) {
        int m = m_base + wave * 16 + v + hf * 8;
        int b = m >> 11;
        int s = m & 2047;
        float val = acc[ns][v] + bv;
        size_t bh = (size_t)(b * NH + h);
        if (which == 0)      Qb[(bh * SEQ + s) * HD + e] = to_bf16(val * 0.125f); // 1/sqrt(64)
        else if (which == 1) Kb[(bh * SEQ + s) * HD + e] = to_bf16(val);
        else                 Vtb[(bh * HD + e) * SEQ + s] = to_bf16(val);
      }
    }
  });
}

// Output projection: d_out[8192,1024] = A @ w_proj + b_proj (f32 out).
__global__ __launch_bounds__(256) void proj_gemm_kernel(
    const __bf16* __restrict__ A, const __bf16* __restrict__ Bt,
    const float* __restrict__ bias, float* __restrict__ out) {
  __shared__ alignas(16) __bf16 sA[2][128 * 32];
  __shared__ alignas(16) __bf16 sB[2][128 * 32];
  const int lane = threadIdx.x & 31;
  const int wave = threadIdx.x >> 5;
  const int hf   = (lane >> 4) & 1;
  const int l16  = lane & 15;
  const int m_base = blockIdx.x * 128;
  const int n_base = blockIdx.y * 128;

  gemm_core(A, Bt, &sA[0][0], &sB[0][0], m_base, n_base, [&](v8f* acc) {
#pragma unroll
    for (int ns = 0; ns < 8; ns++) {
      int ncol = n_base + ns * 16 + l16;
      float bv = bias[ncol];
#pragma unroll
      for (int v = 0; v < 8; v++) {
        int m = m_base + wave * 16 + v + hf * 8;
        out[(size_t)m * DIM_ + ncol] = acc[ns][v] + bv;
      }
    }
  });
}

// ---------------------------------------------------------------------------
// Flash attention: block = 128 queries of one (b,h); 8 waves, wave owns 16 rows.
// Online softmax state (m,l) per row lives in-wave (16-lane shfl reductions).
// K/V tiles double-buffered with async global->LDS copies.
__global__ __launch_bounds__(256) void flash_attn_kernel(
    const __bf16* __restrict__ Qb, const __bf16* __restrict__ Kb,
    const __bf16* __restrict__ Vtb, __bf16* __restrict__ Ob) {
  __shared__ alignas(16) __bf16 sQ[128 * 64];        // Q tile, loaded once
  __shared__ alignas(16) __bf16 sK[2][64 * 64];      // K tile [sk][hd]  == B^T for Q@K^T
  __shared__ alignas(16) __bf16 sV[2][64 * 64];      // V^T tile [hd][sk] == B^T for P@V
  __shared__ alignas(16) __bf16 sP[8 * 16 * 64];     // per-wave P scratch (C->A layout)

  const int tid  = threadIdx.x;
  const int lane = tid & 31;
  const int wave = tid >> 5;
  const int hf   = (lane >> 4) & 1;
  const int l16  = lane & 15;
  const int bh   = blockIdx.x;            // 0..63
  const int q0   = blockIdx.y * 128;
  const size_t qkBase = (size_t)bh * SEQ * HD;
  const size_t vtBase = (size_t)bh * HD * SEQ;

  // stage one 64-key K tile + V^T tile into buffer `buf`
  auto stage_kv = [&](int sk0, int buf) {
    const __bf16* gK = Kb + qkBase + (size_t)sk0 * HD;
#pragma unroll
    for (int i = 0; i < 2; i++) {               // 512 chunks each, 2 per thread
      int c = tid + i * 256;
      copy16_to_lds(&sK[buf][c * 8], gK + c * 8);
    }
#pragma unroll
    for (int i = 0; i < 2; i++) {
      int c = tid + i * 256;
      int e = c >> 3, off = c & 7;              // V^T row e, 8 chunks of 8 per row
      copy16_to_lds(&sV[buf][e * 64 + off * 8],
                    Vtb + vtBase + (size_t)e * SEQ + sk0 + off * 8);
    }
  };

  { // load 128x64 Q tile once (plain loads; overlapped with first K/V stage)
    const uint4* g = reinterpret_cast<const uint4*>(Qb + qkBase + (size_t)q0 * HD);
    uint4* s = reinterpret_cast<uint4*>(sQ);
#pragma unroll
    for (int i = 0; i < 4; i++) s[tid + i * 256] = g[tid + i * 256];
  }
  stage_kv(0, 0);

  v8f zero = {0.f, 0.f, 0.f, 0.f, 0.f, 0.f, 0.f, 0.f};
  float mrow[8], lrow[8];
  v8f oacc[4];
#pragma unroll
  for (int v = 0; v < 8; v++) { mrow[v] = -3.0e38f; lrow[v] = 0.f; }
#pragma unroll
  for (int n = 0; n < 4; n++) oacc[n] = zero;

  copy_fence();
  __syncthreads();

  __bf16* pbuf = sP + wave * 16 * 64;
  const int NT = SEQ / 64;                      // 32 key tiles

  for (int t = 0; t < NT; t++) {
    const int buf = t & 1;
    if (t + 1 < NT) stage_kv((t + 1) * 64, buf ^ 1);   // async prefetch next tile

    // S = Q @ K^T  (16 rows x 64 keys per wave)
    v8f sacc[4];
#pragma unroll
    for (int n = 0; n < 4; n++) sacc[n] = zero;
#pragma unroll
    for (int kk = 0; kk < 64; kk += 32) {
      v16bf aq = load_A16x32(sQ, 64, wave * 16, kk, lane);
#pragma unroll
      for (int n = 0; n < 4; n++) {
        v16bf bk = load_B32x16(sK[buf], 64, n * 16, kk, lane);
        sacc[n] = __builtin_amdgcn_wmma_f32_16x16x32_bf16(
            false, aq, false, bk, (short)0, sacc[n], false, false);
      }
    }

    // online softmax update; write P (bf16) to per-wave LDS scratch
    float alpha[8];
#pragma unroll
    for (int v = 0; v < 8; v++) {
      float mx = fmaxf(fmaxf(sacc[0][v], sacc[1][v]), fmaxf(sacc[2][v], sacc[3][v]));
      mx = fmaxf(mx, __shfl_xor(mx, 1, 32));
      mx = fmaxf(mx, __shfl_xor(mx, 2, 32));
      mx = fmaxf(mx, __shfl_xor(mx, 4, 32));
      mx = fmaxf(mx, __shfl_xor(mx, 8, 32));
      float mnew = fmaxf(mrow[v], mx);
      float a    = __expf(mrow[v] - mnew);
      float rsum = 0.f;
      int row = v + hf * 8;
#pragma unroll
      for (int n = 0; n < 4; n++) {
        float p = __expf(sacc[n][v] - mnew);
        rsum += p;
        pbuf[row * 64 + n * 16 + l16] = to_bf16(p);
      }
      rsum += __shfl_xor(rsum, 1, 32);
      rsum += __shfl_xor(rsum, 2, 32);
      rsum += __shfl_xor(rsum, 4, 32);
      rsum += __shfl_xor(rsum, 8, 32);
      lrow[v]  = lrow[v] * a + rsum;
      mrow[v]  = mnew;
      alpha[v] = a;
    }
    asm volatile("" ::: "memory"); // per-wave LDS is in-order; stop compiler reordering

    // O = diag(alpha) * O + P @ V
#pragma unroll
    for (int n = 0; n < 4; n++) {
      v8f c;
#pragma unroll
      for (int v = 0; v < 8; v++) c[v] = oacc[n][v] * alpha[v];
#pragma unroll
      for (int kk = 0; kk < 64; kk += 32) {
        v16bf ap = load_A16x32(pbuf, 64, 0, kk, lane);
        v16bf bv = load_B32x16(sV[buf], 64, n * 16, kk, lane);
        c = __builtin_amdgcn_wmma_f32_16x16x32_bf16(
            false, ap, false, bv, (short)0, c, false, false);
      }
      oacc[n] = c;
    }
    copy_fence();
    __syncthreads();
  }

  // normalize and write attention output as bf16 [B*S, D] (feeds proj GEMM)
  const int b = bh >> 4, h = bh & 15;
#pragma unroll
  for (int n = 0; n < 4; n++) {
#pragma unroll
    for (int v = 0; v < 8; v++) {
      int row = q0 + wave * 16 + v + hf * 8;
      int col = h * HD + n * 16 + l16;
      float o = oacc[n][v] / lrow[v];
      Ob[(size_t)(b * SEQ + row) * DIM_ + col] = to_bf16(o);
    }
  }
}

// ---------------------------------------------------------------------------
extern "C" void kernel_launch(void* const* d_in, const int* in_sizes, int n_in,
                              void* d_out, int out_size, void* d_ws, size_t ws_size,
                              hipStream_t stream) {
  const float* x      = (const float*)d_in[0];
  const float* w_qkv  = (const float*)d_in[1];
  const float* b_qkv  = (const float*)d_in[2];
  const float* w_proj = (const float*)d_in[3];
  const float* b_proj = (const float*)d_in[4];
  float* out = (float*)d_out;

  // workspace layout (bf16 staging, ~88 MiB total; fits in 192 MB L2)
  char* ws = (char*)d_ws;
  size_t o = 0;
  __bf16* xh     = (__bf16*)(ws + o); o += (size_t)MTOT * DIM_ * 2;          // x in bf16
  __bf16* wqkvT  = (__bf16*)(ws + o); o += (size_t)3 * DIM_ * DIM_ * 2;      // w_qkv^T
  __bf16* wprojT = (__bf16*)(ws + o); o += (size_t)DIM_ * DIM_ * 2;          // w_proj^T
  __bf16* Qb     = (__bf16*)(ws + o); o += (size_t)MTOT * DIM_ * 2;          // [B,H,S,64] (pre-scaled)
  __bf16* Kb     = (__bf16*)(ws + o); o += (size_t)MTOT * DIM_ * 2;          // [B,H,S,64]
  __bf16* Vtb    = (__bf16*)(ws + o); o += (size_t)MTOT * DIM_ * 2;          // [B,H,64,S]
  __bf16* attnO  = (__bf16*)(ws + o); o += (size_t)MTOT * DIM_ * 2;          // [B*S, D]

  const int nx = MTOT * DIM_;
  cvt_bf16_kernel<<<(nx + 255) / 256, 256, 0, stream>>>(x, xh, nx);
  cvt_T_bf16_kernel<<<(3 * DIM_ * DIM_ + 255) / 256, 256, 0, stream>>>(w_qkv, wqkvT, DIM_, 3 * DIM_);
  cvt_T_bf16_kernel<<<(DIM_ * DIM_ + 255) / 256, 256, 0, stream>>>(w_proj, wprojT, DIM_, DIM_);

  qkv_gemm_kernel<<<dim3(MTOT / 128, (3 * DIM_) / 128), 256, 0, stream>>>(
      xh, wqkvT, b_qkv, Qb, Kb, Vtb);

  flash_attn_kernel<<<dim3(BATCH * NH, SEQ / 128), 256, 0, stream>>>(Qb, Kb, Vtb, attnO);

  proj_gemm_kernel<<<dim3(MTOT / 128, DIM_ / 128), 256, 0, stream>>>(
      attnO, wprojT, b_proj, out);
}
